// STGAT_SensorFusion_15891378995373
// MI455X (gfx1250) — compile-verified, matrix operand order checked by reading, not run on hardware
//
#include <hip/hip_runtime.h>
#include <hip/hip_bf16.h>
#include <math.h>

typedef __attribute__((ext_vector_type(16))) _Float16 v16h;
typedef __attribute__((ext_vector_type(8)))  float    v8f;

#define HEADS 4
#define KNN_K 16
#define NN    2048   // nodes per (b,t)
#define BT    32     // B*T
#define DD    64     // feature dim
#define NNODE 65536  // BT*NN
#define NQ    16384  // B*NN (query rows for temporal attention)

// ---------------------------------------------------------------------------
// Weight conversion: f32 -> f16 copies of the three 64x64 GEMM weights.
// ---------------------------------------------------------------------------
__global__ void cvt_w_kernel(const float* __restrict__ gat,
                             const float* __restrict__ qw,
                             const float* __restrict__ kvw,
                             _Float16* __restrict__ g16,
                             _Float16* __restrict__ q16,
                             _Float16* __restrict__ k16) {
    int i = blockIdx.x * 256 + threadIdx.x;
    if (i < DD * DD) {
        g16[i] = (_Float16)gat[i];
        q16[i] = (_Float16)qw[i];
        k16[i] = (_Float16)kvw[i];
    }
}

// ---------------------------------------------------------------------------
// Embedding MLP: x(11) -> 32 (BN+relu) -> 64 (BN+relu). One thread per node.
// Writes h in f32 (needed for the GAT residual) and f16 (for WMMA GEMM).
// ---------------------------------------------------------------------------
__global__ void embed_kernel(const float* __restrict__ x,
                             const float* __restrict__ w1, const float* __restrict__ b1,
                             const float* __restrict__ g1, const float* __restrict__ bb1,
                             const float* __restrict__ m1, const float* __restrict__ v1,
                             const float* __restrict__ w2, const float* __restrict__ b2,
                             const float* __restrict__ g2, const float* __restrict__ bb2,
                             const float* __restrict__ m2, const float* __restrict__ v2,
                             float* __restrict__ h, _Float16* __restrict__ h16) {
    int node = blockIdx.x * 256 + threadIdx.x;      // < 65536
    const float* xi = x + (size_t)node * 11;
    float xv[11];
#pragma unroll
    for (int k = 0; k < 11; ++k) xv[k] = xi[k];
    float h1[32];
#pragma unroll 4
    for (int c = 0; c < 32; ++c) {
        float y = b1[c];
#pragma unroll
        for (int k = 0; k < 11; ++k) y += xv[k] * w1[c * 11 + k];
        float s = g1[c] * rsqrtf(v1[c] + 1e-5f);
        y = (y - m1[c]) * s + bb1[c];
        h1[c] = fmaxf(y, 0.f);
    }
    float*    ho   = h   + (size_t)node * DD;
    _Float16* ho16 = h16 + (size_t)node * DD;
#pragma unroll 2
    for (int c = 0; c < DD; ++c) {
        float y = b2[c];
#pragma unroll
        for (int k = 0; k < 32; ++k) y += h1[k] * w2[c * 32 + k];
        float s = g2[c] * rsqrtf(v2[c] + 1e-5f);
        y = fmaxf((y - m2[c]) * s + bb2[c], 0.f);
        ho[c]   = y;
        ho16[c] = (_Float16)y;
    }
}

// ---------------------------------------------------------------------------
// Fused KNN: one thread per query node, candidates tiled through LDS,
// top-16 maintained as an insertion-sorted register array (never touch HBM
// with the 2048x2048 distance matrix). Same d2 formula as the reference
// (sq_i + sq_j - 2*dot) so values (and ties, broken by lower index) match.
// ---------------------------------------------------------------------------
__global__ void knn_kernel(const float* __restrict__ x, int* __restrict__ idx) {
    __shared__ float sx[256], sy[256], ssq[256];
    int bt = blockIdx.x >> 3;
    int q  = ((blockIdx.x & 7) << 8) + threadIdx.x;
    const float* xb = x + (size_t)bt * NN * 11;
    float qx = xb[q * 11 + 0], qy = xb[q * 11 + 1];
    float qsq = qx * qx + qy * qy;
    float bd[KNN_K]; int bi[KNN_K];
#pragma unroll
    for (int k = 0; k < KNN_K; ++k) { bd[k] = 3.4e38f; bi[k] = 0; }
    for (int c0 = 0; c0 < NN; c0 += 256) {
        int j = c0 + threadIdx.x;
        float cx = xb[j * 11 + 0], cy = xb[j * 11 + 1];
        sx[threadIdx.x] = cx; sy[threadIdx.x] = cy;
        ssq[threadIdx.x] = cx * cx + cy * cy;
        __syncthreads();
        for (int jj = 0; jj < 256; ++jj) {
            float d2 = qsq + ssq[jj] - 2.f * (qx * sx[jj] + qy * sy[jj]);
            if (d2 < bd[KNN_K - 1]) {          // strict: ties keep earlier index
                bd[KNN_K - 1] = d2; bi[KNN_K - 1] = c0 + jj;
#pragma unroll
                for (int p = KNN_K - 1; p > 0; --p) {
                    if (bd[p] < bd[p - 1]) {
                        float td = bd[p]; bd[p] = bd[p - 1]; bd[p - 1] = td;
                        int   ti = bi[p]; bi[p] = bi[p - 1]; bi[p - 1] = ti;
                    }
                }
            }
        }
        __syncthreads();
    }
    int* op = idx + ((size_t)bt * NN + q) * KNN_K;
#pragma unroll
    for (int k = 0; k < KNN_K; ++k) op[k] = bi[k];
}

// ---------------------------------------------------------------------------
// WMMA fragment loader (16-bit, ISA 16x32 A layout; also used for B=W^T since
// W is row-major [Ncols][K]). lanes 0-15: rows M, K base 0/16; lanes 16-31:
// same rows, K base 8/24.
// ---------------------------------------------------------------------------
__device__ __forceinline__ v16h load_frag16(const _Float16* __restrict__ base,
                                            int row0, int k0, int lane, int remap) {
    int r = row0 + (lane & 15);
    if (remap) r = r + 6144 * ((r >> 11) + 1);   // Q rows: (4b+3)*2048+n
    const _Float16* p = base + (size_t)r * DD + k0 + ((lane >> 4) ? 8 : 0);
    v16h f;
#pragma unroll
    for (int i = 0; i < 4; ++i) { f[2 * i]     = p[2 * i];      f[2 * i + 1]     = p[2 * i + 1]; }
#pragma unroll
    for (int i = 0; i < 4; ++i) { f[8 + 2 * i] = p[16 + 2 * i]; f[8 + 2 * i + 1] = p[16 + 2 * i + 1]; }
    return f;
}

// ---------------------------------------------------------------------------
// WMMA GEMM: C[M x 64] = A[M x 64](f16) * W^T (W f16 [64][64]) + bias.
// One wave per 16x64 output STRIP: each A fragment is loaded once and reused
// against all 4 column tiles of W (8 KB, L0-resident) -> 8 WMMAs per wave,
// 4 v8f accumulators, 4x less A-side VMEM traffic than tile-per-wave.
// Grids sized exactly; no divergent branches -> EXEC all-ones around WMMA.
// ---------------------------------------------------------------------------
__global__ void wmma_gemm64_kernel(const _Float16* __restrict__ A,
                                   const _Float16* __restrict__ W,
                                   const float* __restrict__ bias,
                                   float* __restrict__ C,
                                   int remap) {
    int wave = (blockIdx.x * 256 + threadIdx.x) >> 5;
    int lane = threadIdx.x & 31;
    int row0 = wave << 4;                    // 16-row strip
    v8f acc0 = {}, acc1 = {}, acc2 = {}, acc3 = {};
#pragma unroll
    for (int k0 = 0; k0 < DD; k0 += 32) {
        v16h a  = load_frag16(A, row0, k0, lane, remap);
        v16h b0 = load_frag16(W,  0, k0, lane, 0);
        acc0 = __builtin_amdgcn_wmma_f32_16x16x32_f16(false, a, false, b0, (short)0, acc0, false, false);
        v16h b1 = load_frag16(W, 16, k0, lane, 0);
        acc1 = __builtin_amdgcn_wmma_f32_16x16x32_f16(false, a, false, b1, (short)0, acc1, false, false);
        v16h b2 = load_frag16(W, 32, k0, lane, 0);
        acc2 = __builtin_amdgcn_wmma_f32_16x16x32_f16(false, a, false, b2, (short)0, acc2, false, false);
        v16h b3 = load_frag16(W, 48, k0, lane, 0);
        acc3 = __builtin_amdgcn_wmma_f32_16x16x32_f16(false, a, false, b3, (short)0, acc3, false, false);
    }
    int colb  = lane & 15;
    int rbase = row0 + ((lane >> 4) ? 8 : 0);
    float bv0 = 0.f, bv1 = 0.f, bv2 = 0.f, bv3 = 0.f;
    if (bias) {
        bv0 = bias[colb]; bv1 = bias[16 + colb];
        bv2 = bias[32 + colb]; bv3 = bias[48 + colb];
    }
#pragma unroll
    for (int j = 0; j < 8; ++j) {
        float* cp = C + (size_t)(rbase + j) * DD + colb;
        cp[0]  = acc0[j] + bv0;
        cp[16] = acc1[j] + bv1;
        cp[32] = acc2[j] + bv2;
        cp[48] = acc3[j] + bv3;
    }
}

// ---------------------------------------------------------------------------
// GAT attention: one thread per (row, head). Scores over 16 gathered
// neighbors (hw is L2-resident), leaky-relu, softmax over K, weighted sum,
// residual + relu -> h_sp (f16, consumed only by WMMA GEMMs).
// ---------------------------------------------------------------------------
__global__ void gat_kernel(const float* __restrict__ hw, const float* __restrict__ h,
                           const int* __restrict__ idx, const float* __restrict__ att,
                           _Float16* __restrict__ hsp16) {
    int t    = blockIdx.x * 256 + threadIdx.x;   // 262144
    int head = t & 3;
    int row  = t >> 2;                           // bt*2048+n
    int btbase = (row >> 11) << 11;
    const int hd = DD / HEADS;                   // 16
    const float* hwr = hw + (size_t)row * DD + head * hd;
    const float* as  = att + head * 2 * hd;
    const float* an  = as + hd;
    float ss = 0.f;
#pragma unroll
    for (int d = 0; d < hd; ++d) ss += hwr[d] * as[d];
    const int* nb = idx + (size_t)row * KNN_K;
    float sc[KNN_K]; float mx = -3.4e38f;
#pragma unroll
    for (int k = 0; k < KNN_K; ++k) {
        const float* hn = hw + (size_t)(btbase + nb[k]) * DD + head * hd;
        float s = 0.f;
#pragma unroll
        for (int d = 0; d < hd; ++d) s += hn[d] * an[d];
        s += ss;
        s = (s >= 0.f) ? s : 0.2f * s;           // leaky_relu(0.2)
        sc[k] = s; mx = fmaxf(mx, s);
    }
    float sum = 0.f;
#pragma unroll
    for (int k = 0; k < KNN_K; ++k) { sc[k] = __expf(sc[k] - mx); sum += sc[k]; }
    float inv = 1.f / sum;
    float out[16];
#pragma unroll
    for (int d = 0; d < hd; ++d) out[d] = 0.f;
#pragma unroll
    for (int k = 0; k < KNN_K; ++k) {
        const float* hn = hw + (size_t)(btbase + nb[k]) * DD + head * hd;
        float a = sc[k] * inv;
#pragma unroll
        for (int d = 0; d < hd; ++d) out[d] += a * hn[d];
    }
    const float* hr = h + (size_t)row * DD + head * hd;
    _Float16* o = hsp16 + (size_t)row * DD + head * hd;
#pragma unroll
    for (int d = 0; d < hd; ++d)
        o[d] = (_Float16)fmaxf(out[d] + hr[d], 0.f);
}

// ---------------------------------------------------------------------------
// Temporal attention over T=4 + both output heads, fused. One thread per
// (b, n). KV rows live in L2.
// ---------------------------------------------------------------------------
__device__ __forceinline__ float softplus1(float v) {
    return ((v > 20.f) ? v : log1pf(__expf(v))) + 1e-6f;
}

__device__ __forceinline__ void mlp_head4(const float* __restrict__ hf,
                                          const float* __restrict__ w1, const float* __restrict__ b1,
                                          const float* __restrict__ w2, const float* __restrict__ b2,
                                          float* __restrict__ o) {
    float o0 = b2[0], o1 = b2[1], o2 = b2[2], o3 = b2[3];
#pragma unroll 4
    for (int w = 0; w < 32; ++w) {
        float s = b1[w];
#pragma unroll
        for (int d = 0; d < DD; ++d) s += hf[d] * w1[w * DD + d];
        s = fmaxf(s, 0.f);
        o0 += s * w2[0 * 32 + w]; o1 += s * w2[1 * 32 + w];
        o2 += s * w2[2 * 32 + w]; o3 += s * w2[3 * 32 + w];
    }
    o[0] = o0; o[1] = o1; o[2] = o2; o[3] = o3;
}

__global__ void fuse_head_kernel(const float* __restrict__ Q, const float* __restrict__ KV,
                                 const float* __restrict__ lw1, const float* __restrict__ lb1,
                                 const float* __restrict__ lw2, const float* __restrict__ lb2,
                                 const float* __restrict__ rw1, const float* __restrict__ rb1,
                                 const float* __restrict__ rw2, const float* __restrict__ rb2,
                                 float* __restrict__ out) {
    int t = blockIdx.x * 256 + threadIdx.x;    // 16384
    int b = t >> 11, n = t & 2047;
    const float* q = Q + (size_t)t * DD;
    float qv[DD];
#pragma unroll
    for (int d = 0; d < DD; ++d) qv[d] = q[d];
    float sc[4]; float mx = -3.4e38f;
#pragma unroll
    for (int tt = 0; tt < 4; ++tt) {
        const float* kv = KV + (size_t)(((b * 4 + tt) << 11) + n) * DD;
        float s = 0.f;
#pragma unroll
        for (int d = 0; d < DD; ++d) s += qv[d] * kv[d];
        s *= 0.125f;                           // 1/sqrt(64)
        sc[tt] = s; mx = fmaxf(mx, s);
    }
    float sum = 0.f;
#pragma unroll
    for (int tt = 0; tt < 4; ++tt) { sc[tt] = __expf(sc[tt] - mx); sum += sc[tt]; }
    float inv = 1.f / sum;
    float hf[DD];
#pragma unroll
    for (int d = 0; d < DD; ++d) hf[d] = 0.f;
#pragma unroll
    for (int tt = 0; tt < 4; ++tt) {
        const float* kv = KV + (size_t)(((b * 4 + tt) << 11) + n) * DD;
        float w = sc[tt] * inv;
#pragma unroll
        for (int d = 0; d < DD; ++d) hf[d] += w * kv[d];
    }
    float ol[4], orr[4];
    mlp_head4(hf, lw1, lb1, lw2, lb2, ol);
    mlp_head4(hf, rw1, rb1, rw2, rb2, orr);
#pragma unroll
    for (int j = 0; j < 2; ++j) {
        out[t * 2 + j]          = ol[j];               // lidar_mu
        out[32768 + t * 2 + j]  = softplus1(ol[2 + j]);// lidar_sigma
        out[65536 + t * 2 + j]  = orr[j];              // radar_mu
        out[98304 + t * 2 + j]  = softplus1(orr[2 + j]);// radar_sigma
    }
}

// ---------------------------------------------------------------------------
extern "C" void kernel_launch(void* const* d_in, const int* in_sizes, int n_in,
                              void* d_out, int out_size, void* d_ws, size_t ws_size,
                              hipStream_t stream) {
    const float* x      = (const float*)d_in[0];
    const float* emb_w1 = (const float*)d_in[1];
    const float* emb_b1 = (const float*)d_in[2];
    const float* bn1_g  = (const float*)d_in[3];
    const float* bn1_b  = (const float*)d_in[4];
    const float* bn1_m  = (const float*)d_in[5];
    const float* bn1_v  = (const float*)d_in[6];
    const float* emb_w2 = (const float*)d_in[7];
    const float* emb_b2 = (const float*)d_in[8];
    const float* bn2_g  = (const float*)d_in[9];
    const float* bn2_b  = (const float*)d_in[10];
    const float* bn2_m  = (const float*)d_in[11];
    const float* bn2_v  = (const float*)d_in[12];
    const float* gat_W  = (const float*)d_in[13];
    const float* att    = (const float*)d_in[14];
    const float* q_w    = (const float*)d_in[15];
    const float* q_b    = (const float*)d_in[16];
    const float* kv_w   = (const float*)d_in[17];
    const float* kv_b   = (const float*)d_in[18];
    const float* lw1    = (const float*)d_in[19];
    const float* lb1    = (const float*)d_in[20];
    const float* lw2    = (const float*)d_in[21];
    const float* lb2    = (const float*)d_in[22];
    const float* rw1    = (const float*)d_in[23];
    const float* rb1    = (const float*)d_in[24];
    const float* rw2    = (const float*)d_in[25];
    const float* rb2    = (const float*)d_in[26];
    float* out = (float*)d_out;

    // Workspace layout (all offsets 256B-aligned).
    char* ws = (char*)d_ws;
    size_t off = 0;
    float*    h     = (float*)(ws + off);    off += (size_t)NNODE * DD * 4;  // 16 MB
    _Float16* h16   = (_Float16*)(ws + off); off += (size_t)NNODE * DD * 2;  //  8 MB
    float*    hw    = (float*)(ws + off);    off += (size_t)NNODE * DD * 4;  // 16 MB
    int*      idx   = (int*)(ws + off);      off += (size_t)NNODE * KNN_K * 4; // 4 MB
    _Float16* hsp16 = (_Float16*)(ws + off); off += (size_t)NNODE * DD * 2;  //  8 MB
    float*    Qr    = (float*)(ws + off);    off += (size_t)NQ * DD * 4;     //  4 MB
    float*    KVr   = (float*)(ws + off);    off += (size_t)NNODE * DD * 4;  // 16 MB
    _Float16* g16   = (_Float16*)(ws + off); off += (size_t)DD * DD * 2;
    _Float16* q16   = (_Float16*)(ws + off); off += (size_t)DD * DD * 2;
    _Float16* k16   = (_Float16*)(ws + off); off += (size_t)DD * DD * 2;

    cvt_w_kernel<<<16, 256, 0, stream>>>(gat_W, q_w, kv_w, g16, q16, k16);

    embed_kernel<<<NNODE / 256, 256, 0, stream>>>(
        x, emb_w1, emb_b1, bn1_g, bn1_b, bn1_m, bn1_v,
        emb_w2, emb_b2, bn2_g, bn2_b, bn2_m, bn2_v, h, h16);

    knn_kernel<<<BT * (NN / 256), 256, 0, stream>>>(x, idx);

    // hw = h @ gat_W.T : 4096 row strips, 8 waves/block -> 512 blocks
    wmma_gemm64_kernel<<<512, 256, 0, stream>>>(h16, g16, nullptr, hw, 0);

    gat_kernel<<<(NNODE * HEADS) / 256, 256, 0, stream>>>(hw, h, idx, att, hsp16);

    // KV = h_sp @ kv_w.T + kv_b : all 65536 rows -> 4096 strips
    wmma_gemm64_kernel<<<512, 256, 0, stream>>>(hsp16, k16, kv_b, KVr, 0);
    // Q  = h_sp[t==3 rows] @ q_w.T + q_b : 16384 rows -> 1024 strips
    wmma_gemm64_kernel<<<128, 256, 0, stream>>>(hsp16, q16, q_b, Qr, 1);

    fuse_head_kernel<<<NQ / 256, 256, 0, stream>>>(
        Qr, KVr, lw1, lb1, lw2, lb2, rw1, rb1, rw2, rb2, out);

    (void)in_sizes; (void)n_in; (void)out_size; (void)ws_size;
}
